// TransformerXLBlock_45543833207176
// MI455X (gfx1250) — compile-verified, matrix operand order checked by reading, not run on hardware
//
#include <hip/hip_runtime.h>

// ---------------------------------------------------------------------------
// Problem constants (TransformerXL block, fp32 reference)
// ---------------------------------------------------------------------------
constexpr int BB   = 4;
constexpr int TT   = 1024;
constexpr int MEM  = 1024;
constexpr int DD   = 1024;
constexpr int HH   = 16;
constexpr int DH   = 64;
constexpr int LL   = TT + MEM;   // 2048
constexpr int DFF  = 4096;

typedef __bf16 v16bf __attribute__((ext_vector_type(16)));
typedef float  v8f   __attribute__((ext_vector_type(8)));

union FragBF {
    uint4 u[2];   // two 16-byte runs of 8 bf16 each
    v16bf v;
};

// ---------------------------------------------------------------------------
// Generic batched WMMA GEMM with gfx1250 async global->LDS double buffering:
//   C[z][m,n] (f32, ldc) = A[z][m,k] (bf16 row-major, lda)
//                        x Bt[z % btMod][n,k] (bf16 row-major = B^T, ldb)
// Block tile 128x64, 8 waves, each wave a 32x32 sub-tile (2x2 of 16x16).
// K mult of 32; M mult of 128; N mult of 64 (true for every call site).
// Tiles are staged with global_load_async_to_lds_b128 (ASYNCcnt), ping-pong
// buffered so tile kk+1 copies overlap tile kk WMMAs.
// ---------------------------------------------------------------------------
#define MBLK 128
#define NBLK 64
#define KBLK 32

__device__ __forceinline__ void async_b128(unsigned lds_off, unsigned long long gaddr)
{
    asm volatile("global_load_async_to_lds_b128 %0, %1, off"
                 :: "v"(lds_off), "v"(gaddr) : "memory");
}

__global__ __launch_bounds__(256)
void gemm_bf16_wmma(const __bf16* __restrict__ A, int lda, long long aBatch,
                    const __bf16* __restrict__ Bt, int ldb, long long btBatch, int btMod,
                    float* __restrict__ C, int ldc, long long cBatch,
                    int K)
{
    __shared__ __bf16 sA[2][MBLK][KBLK];
    __shared__ __bf16 sB[2][NBLK][KBLK];

    const int z = blockIdx.z;
    const __bf16* Ab = A  + (long long)z * aBatch;
    const __bf16* Bb = Bt + (long long)(z % btMod) * btBatch;
    float*        Cb = C  + (long long)z * cBatch;

    const int m0   = blockIdx.y * MBLK;
    const int n0   = blockIdx.x * NBLK;
    const int tid  = threadIdx.x;
    const int wave = tid >> 5;
    const int lane = tid & 31;
    const int wm   = (wave & 3) * 32;   // 4 waves along M
    const int wn   = (wave >> 2) * 32;  // 2 waves along N

    // 16B-chunk staging coordinates: A is 512 chunks (2/thread), B is 256 (1/thread)
    const int car = tid >> 2;          // chunk row 0..63
    const int cac = (tid & 3) * 8;     // chunk col (bf16 elements)

    const __bf16* gA0 = Ab + (long long)(m0 + car) * lda + cac;
    const __bf16* gA1 = Ab + (long long)(m0 + 64 + car) * lda + cac;
    const __bf16* gB0 = Bb + (long long)(n0 + car) * ldb + cac;

    auto issue_tile = [&](int kk, int buf) {
        async_b128((unsigned)(size_t)&sA[buf][car][cac],
                   (unsigned long long)(size_t)(gA0 + kk));
        async_b128((unsigned)(size_t)&sA[buf][64 + car][cac],
                   (unsigned long long)(size_t)(gA1 + kk));
        async_b128((unsigned)(size_t)&sB[buf][car][cac],
                   (unsigned long long)(size_t)(gB0 + kk));
    };

    v8f acc[2][2] = {};

    // Fragment addressing per CDNA5 16-bit A-matrix layout:
    // lane l holds row (l&15); K runs base=(l>=16?8:0)+{0..7} and base+16+{0..7}
    const int fr = lane & 15;
    const int fk = (lane >> 4) * 8;

    issue_tile(0, 0);
    int buf = 0;
    for (int kk = 0; kk < K; kk += KBLK, buf ^= 1) {
        asm volatile("s_wait_asynccnt 0x0" ::: "memory");  // own copies done
        __syncthreads();                                   // everyone's copies done
        if (kk + KBLK < K) issue_tile(kk + KBLK, buf ^ 1); // overlap next tile

        FragBF a0, a1, b0, b1;
        a0.u[0] = *(const uint4*)&sA[buf][wm + fr][fk];
        a0.u[1] = *(const uint4*)&sA[buf][wm + fr][fk + 16];
        a1.u[0] = *(const uint4*)&sA[buf][wm + 16 + fr][fk];
        a1.u[1] = *(const uint4*)&sA[buf][wm + 16 + fr][fk + 16];
        b0.u[0] = *(const uint4*)&sB[buf][wn + fr][fk];
        b0.u[1] = *(const uint4*)&sB[buf][wn + fr][fk + 16];
        b1.u[0] = *(const uint4*)&sB[buf][wn + 16 + fr][fk];
        b1.u[1] = *(const uint4*)&sB[buf][wn + 16 + fr][fk + 16];

        acc[0][0] = __builtin_amdgcn_wmma_f32_16x16x32_bf16(false, a0.v, false, b0.v,
                                                            (short)0, acc[0][0], false, false);
        acc[0][1] = __builtin_amdgcn_wmma_f32_16x16x32_bf16(false, a0.v, false, b1.v,
                                                            (short)0, acc[0][1], false, false);
        acc[1][0] = __builtin_amdgcn_wmma_f32_16x16x32_bf16(false, a1.v, false, b0.v,
                                                            (short)0, acc[1][0], false, false);
        acc[1][1] = __builtin_amdgcn_wmma_f32_16x16x32_bf16(false, a1.v, false, b1.v,
                                                            (short)0, acc[1][1], false, false);
        // no trailing barrier: buffer `buf` is only rewritten after the next
        // iteration's wait+barrier, by which time all ds_loads have returned
    }

    // C layout: VGPR r holds M = r + (lane>=16 ? 8 : 0), N = lane & 15
    const int cn    = lane & 15;
    const int rbase = (lane >> 4) * 8;
    for (int i = 0; i < 2; ++i)
        for (int j = 0; j < 2; ++j)
            for (int r = 0; r < 8; ++r) {
                int row = m0 + wm + i * 16 + rbase + r;
                int col = n0 + wn + j * 16 + cn;
                Cb[(long long)row * ldc + col] = acc[i][j][r];
            }
}

// ---------------------------------------------------------------------------
// Elementwise / packing kernels
// ---------------------------------------------------------------------------
__global__ void pack_wt_kernel(const float* __restrict__ W, __bf16* __restrict__ Wt,
                               int K, int N)
{
    long long idx = (long long)blockIdx.x * blockDim.x + threadIdx.x;
    long long total = (long long)K * N;
    if (idx >= total) return;
    int k = (int)(idx / N);
    int n = (int)(idx % N);
    Wt[(long long)n * K + k] = (__bf16)W[idx];
}

__global__ void mem_to_cat_kernel(const float* __restrict__ mem, __bf16* __restrict__ cat)
{
    long long idx = (long long)blockIdx.x * blockDim.x + threadIdx.x;
    long long total = (long long)BB * MEM * DD;
    if (idx >= total) return;
    int b = (int)(idx / ((long long)MEM * DD));
    long long r = idx % ((long long)MEM * DD);
    cat[(long long)b * LL * DD + r] = (__bf16)mem[idx];
}

__global__ __launch_bounds__(256)
void ln1_to_cat_kernel(const float* __restrict__ x, const float* __restrict__ g,
                       const float* __restrict__ beta, __bf16* __restrict__ cat)
{
    const int rid = blockIdx.x;          // b*T + t
    const int b = rid / TT, t = rid % TT;
    const float* xr = x + (long long)rid * DD;
    __shared__ float red[256];
    float s = 0.f;
    for (int d = threadIdx.x; d < DD; d += 256) s += xr[d];
    red[threadIdx.x] = s; __syncthreads();
    for (int o = 128; o > 0; o >>= 1) {
        if (threadIdx.x < o) red[threadIdx.x] += red[threadIdx.x + o];
        __syncthreads();
    }
    float mu = red[0] / DD; __syncthreads();
    float v = 0.f;
    for (int d = threadIdx.x; d < DD; d += 256) { float u = xr[d] - mu; v += u * u; }
    red[threadIdx.x] = v; __syncthreads();
    for (int o = 128; o > 0; o >>= 1) {
        if (threadIdx.x < o) red[threadIdx.x] += red[threadIdx.x + o];
        __syncthreads();
    }
    float rstd = rsqrtf(red[0] / DD + 1e-5f);
    __bf16* dst = cat + ((long long)b * LL + MEM + t) * DD;
    for (int d = threadIdx.x; d < DD; d += 256)
        dst[d] = (__bf16)((xr[d] - mu) * rstd * g[d] + beta[d]);
}

__global__ void pos_emb_kernel(__bf16* __restrict__ pe)
{
    long long idx = (long long)blockIdx.x * blockDim.x + threadIdx.x;
    long long total = (long long)LL * (DD / 2);
    if (idx >= total) return;
    int l = (int)(idx / (DD / 2));
    int i = (int)(idx % (DD / 2));
    float pos = (float)(LL - 1 - l);
    float inv = powf(10000.f, -(2.f * i) / (float)DD);
    float s = pos * inv;
    pe[(long long)l * DD + i]          = (__bf16)sinf(s);
    pe[(long long)l * DD + DD / 2 + i] = (__bf16)cosf(s);
}

__global__ void qbias_kernel(const float* __restrict__ q, const float* __restrict__ bu,
                             const float* __restrict__ bv,
                             __bf16* __restrict__ qu, __bf16* __restrict__ qv)
{
    long long idx = (long long)blockIdx.x * blockDim.x + threadIdx.x;
    long long total = (long long)BB * TT * DD;
    if (idx >= total) return;
    int d = (int)(idx % DH);
    int t = (int)((idx / DH) % TT);
    int h = (int)((idx / ((long long)DH * TT)) % HH);
    int b = (int)(idx / ((long long)DH * TT * HH));
    float s = q[((long long)b * TT + t) * DD + h * DH + d];
    qu[idx] = (__bf16)(s + bu[h * DH + d]);
    qv[idx] = (__bf16)(s + bv[h * DH + d]);
}

__global__ void kv_pack_kernel(const float* __restrict__ kf, const float* __restrict__ vf,
                               __bf16* __restrict__ kb, __bf16* __restrict__ vT)
{
    long long idx = (long long)blockIdx.x * blockDim.x + threadIdx.x;
    long long total = (long long)BB * LL * DD;
    if (idx >= total) return;
    int d = (int)(idx % DH);
    int l = (int)((idx / DH) % LL);
    int h = (int)((idx / ((long long)DH * LL)) % HH);
    int b = (int)(idx / ((long long)DH * LL * HH));
    long long src = ((long long)b * LL + l) * DD + h * DH + d;
    kb[idx] = (__bf16)kf[src];
    vT[(((long long)b * HH + h) * DH + d) * LL + l] = (__bf16)vf[src];
}

__global__ void r_pack_kernel(const float* __restrict__ rf, __bf16* __restrict__ rb)
{
    long long idx = (long long)blockIdx.x * blockDim.x + threadIdx.x;
    long long total = (long long)HH * LL * DH;
    if (idx >= total) return;
    int d = (int)(idx % DH);
    int l = (int)((idx / DH) % LL);
    int h = (int)(idx / ((long long)DH * LL));
    rb[idx] = (__bf16)rf[(long long)l * DD + h * DH + d];
}

// softmax((content + rel_shift(rel)) * scale); writes bf16 attn in-place
// over the fp32 score row (attn row stride = 2*L bf16 elements).
__global__ __launch_bounds__(256)
void softmax_shift_kernel(float* __restrict__ S, const float* __restrict__ R)
{
    const long long rid = blockIdx.x;           // bh*T + i
    const int i = (int)(rid % TT);
    const long long bh = rid / TT;
    float* Srow = S + (bh * TT + i) * (long long)LL;
    const float* Rb = R + bh * (long long)TT * LL;
    const float scale = 0.125f;                 // 1/sqrt(dh)

    float vals[LL / 256];
    float mx = -3.0e38f;
    for (int jj = 0; jj < LL / 256; ++jj) {
        int j = threadIdx.x + jj * 256;
        long long gidx = (long long)i * LL + j + TT;
        int sr = (int)(gidx / (LL + 1));
        int sc = (int)(gidx % (LL + 1));
        float rv = (sc == 0) ? 0.f : Rb[(long long)sr * LL + (sc - 1)];
        float sv = (Srow[j] + rv) * scale;
        vals[jj] = sv;
        mx = fmaxf(mx, sv);
    }
    __shared__ float red[256];
    red[threadIdx.x] = mx; __syncthreads();
    for (int o = 128; o > 0; o >>= 1) {
        if (threadIdx.x < o) red[threadIdx.x] = fmaxf(red[threadIdx.x], red[threadIdx.x + o]);
        __syncthreads();
    }
    mx = red[0]; __syncthreads();
    float se = 0.f;
    for (int jj = 0; jj < LL / 256; ++jj) { vals[jj] = expf(vals[jj] - mx); se += vals[jj]; }
    red[threadIdx.x] = se; __syncthreads();
    for (int o = 128; o > 0; o >>= 1) {
        if (threadIdx.x < o) red[threadIdx.x] += red[threadIdx.x + o];
        __syncthreads();
    }
    float inv = 1.f / red[0]; __syncthreads();
    __bf16* arow = (__bf16*)Srow;
    for (int jj = 0; jj < LL / 256; ++jj)
        arow[threadIdx.x + jj * 256] = (__bf16)(vals[jj] * inv);
}

__global__ void merge_heads_kernel(const float* __restrict__ O, __bf16* __restrict__ AO)
{
    long long idx = (long long)blockIdx.x * blockDim.x + threadIdx.x;
    long long total = (long long)BB * TT * DD;
    if (idx >= total) return;
    int co = (int)(idx % DD);
    int t  = (int)((idx / DD) % TT);
    int b  = (int)(idx / ((long long)DD * TT));
    int h  = co / DH, dd = co % DH;
    AO[idx] = (__bf16)O[(((long long)b * HH + h) * TT + t) * DH + dd];
}

__global__ __launch_bounds__(256)
void add_ln2_kernel(const float* __restrict__ x, const float* __restrict__ proj,
                    const float* __restrict__ g, const float* __restrict__ beta,
                    float* __restrict__ x2, __bf16* __restrict__ xf)
{
    const int rid = blockIdx.x;
    const float* xr = x + (long long)rid * DD;
    const float* pr = proj + (long long)rid * DD;
    float* x2r = x2 + (long long)rid * DD;
    __bf16* xfr = xf + (long long)rid * DD;
    __shared__ float red[256];
    float s = 0.f;
    for (int d = threadIdx.x; d < DD; d += 256) {
        float v = xr[d] + pr[d];
        x2r[d] = v;
        s += v;
    }
    red[threadIdx.x] = s; __syncthreads();
    for (int o = 128; o > 0; o >>= 1) {
        if (threadIdx.x < o) red[threadIdx.x] += red[threadIdx.x + o];
        __syncthreads();
    }
    float mu = red[0] / DD; __syncthreads();
    float v = 0.f;
    for (int d = threadIdx.x; d < DD; d += 256) { float u = x2r[d] - mu; v += u * u; }
    red[threadIdx.x] = v; __syncthreads();
    for (int o = 128; o > 0; o >>= 1) {
        if (threadIdx.x < o) red[threadIdx.x] += red[threadIdx.x + o];
        __syncthreads();
    }
    float rstd = rsqrtf(red[0] / DD + 1e-5f);
    for (int d = threadIdx.x; d < DD; d += 256)
        xfr[d] = (__bf16)((x2r[d] - mu) * rstd * g[d] + beta[d]);
}

__global__ void gelu_kernel(const float* __restrict__ ff1, const float* __restrict__ b1,
                            __bf16* __restrict__ gb)
{
    long long idx = (long long)blockIdx.x * blockDim.x + threadIdx.x;
    long long total = (long long)BB * TT * DFF;
    if (idx >= total) return;
    int c = (int)(idx % DFF);
    float v = ff1[idx] + b1[c];
    gb[idx] = (__bf16)(0.5f * v * (1.f + erff(v * 0.70710678118654752f)));
}

__global__ void final_kernel(const float* __restrict__ x, const float* __restrict__ x2,
                             const float* __restrict__ ff2, const float* __restrict__ b2,
                             float* __restrict__ y, float* __restrict__ nm)
{
    long long idx = (long long)blockIdx.x * blockDim.x + threadIdx.x;
    long long total = (long long)BB * TT * DD;
    if (idx >= total) return;
    int d = (int)(idx % DD);
    y[idx]  = x2[idx] + ff2[idx] + b2[d];
    nm[idx] = x[idx];                      // new_mem == residual (T == MEM_LEN)
}

// ---------------------------------------------------------------------------
// Host-side orchestration
// ---------------------------------------------------------------------------
static inline char* ws_take(char*& p, size_t bytes)
{
    char* r = p;
    p += (bytes + 255) & ~size_t(255);
    return r;
}

extern "C" void kernel_launch(void* const* d_in, const int* in_sizes, int n_in,
                              void* d_out, int out_size, void* d_ws, size_t ws_size,
                              hipStream_t stream)
{
    (void)in_sizes; (void)n_in; (void)out_size; (void)ws_size;

    const float* x     = (const float*)d_in[0];
    const float* mem   = (const float*)d_in[1];
    const float* Wq    = (const float*)d_in[2];
    const float* Wk    = (const float*)d_in[3];
    const float* Wv    = (const float*)d_in[4];
    const float* Wr    = (const float*)d_in[5];
    const float* Wo    = (const float*)d_in[6];
    const float* bu    = (const float*)d_in[7];
    const float* bv    = (const float*)d_in[8];
    const float* ln1g  = (const float*)d_in[9];
    const float* ln1b  = (const float*)d_in[10];
    const float* ln2g  = (const float*)d_in[11];
    const float* ln2b  = (const float*)d_in[12];
    const float* W1    = (const float*)d_in[13];
    const float* b1    = (const float*)d_in[14];
    const float* W2    = (const float*)d_in[15];
    const float* b2    = (const float*)d_in[16];

    float* y  = (float*)d_out;
    float* nm = y + (size_t)BB * TT * DD;

    char* p = (char*)d_ws;
    __bf16* cat   = (__bf16*)ws_take(p, sizeof(__bf16) * (size_t)BB * LL * DD);
    __bf16* Wqt   = (__bf16*)ws_take(p, sizeof(__bf16) * (size_t)DD * DD);
    __bf16* Wkt   = (__bf16*)ws_take(p, sizeof(__bf16) * (size_t)DD * DD);
    __bf16* Wvt   = (__bf16*)ws_take(p, sizeof(__bf16) * (size_t)DD * DD);
    __bf16* Wrt   = (__bf16*)ws_take(p, sizeof(__bf16) * (size_t)DD * DD);
    __bf16* Wot   = (__bf16*)ws_take(p, sizeof(__bf16) * (size_t)DD * DD);
    __bf16* W1t   = (__bf16*)ws_take(p, sizeof(__bf16) * (size_t)DD * DFF);
    __bf16* W2t   = (__bf16*)ws_take(p, sizeof(__bf16) * (size_t)DFF * DD);
    __bf16* pe    = (__bf16*)ws_take(p, sizeof(__bf16) * (size_t)LL * DD);
    float*  qf    = (float*) ws_take(p, sizeof(float)  * (size_t)BB * TT * DD);
    float*  kf    = (float*) ws_take(p, sizeof(float)  * (size_t)BB * LL * DD);
    float*  vf    = (float*) ws_take(p, sizeof(float)  * (size_t)BB * LL * DD);
    float*  rf    = (float*) ws_take(p, sizeof(float)  * (size_t)LL * DD);
    __bf16* qu    = (__bf16*)ws_take(p, sizeof(__bf16) * (size_t)BB * TT * DD);
    __bf16* qv    = (__bf16*)ws_take(p, sizeof(__bf16) * (size_t)BB * TT * DD);
    __bf16* kb    = (__bf16*)ws_take(p, sizeof(__bf16) * (size_t)BB * LL * DD);
    __bf16* vT    = (__bf16*)ws_take(p, sizeof(__bf16) * (size_t)BB * LL * DD);
    __bf16* rb    = (__bf16*)ws_take(p, sizeof(__bf16) * (size_t)LL * DD);
    float*  S     = (float*) ws_take(p, sizeof(float)  * (size_t)BB * HH * TT * LL);
    float*  R     = (float*) ws_take(p, sizeof(float)  * (size_t)BB * HH * TT * LL);
    float*  Oaw   = (float*) ws_take(p, sizeof(float)  * (size_t)BB * TT * DD);
    __bf16* AO    = (__bf16*)ws_take(p, sizeof(__bf16) * (size_t)BB * TT * DD);
    float*  proj  = (float*) ws_take(p, sizeof(float)  * (size_t)BB * TT * DD);
    float*  x2    = (float*) ws_take(p, sizeof(float)  * (size_t)BB * TT * DD);
    __bf16* xf    = (__bf16*)ws_take(p, sizeof(__bf16) * (size_t)BB * TT * DD);
    float*  ff1   = (float*) ws_take(p, sizeof(float)  * (size_t)BB * TT * DFF);
    __bf16* gb    = (__bf16*)ws_take(p, sizeof(__bf16) * (size_t)BB * TT * DFF);
    float*  ff2   = (float*) ws_take(p, sizeof(float)  * (size_t)BB * TT * DD);

    auto gcd = [](long long n) { return dim3((unsigned)((n + 255) / 256)); };

    // --- weight packing (W [K,N] fp32 -> W^T [N,K] bf16) ---
    pack_wt_kernel<<<gcd((long long)DD * DD), 256, 0, stream>>>(Wq, Wqt, DD, DD);
    pack_wt_kernel<<<gcd((long long)DD * DD), 256, 0, stream>>>(Wk, Wkt, DD, DD);
    pack_wt_kernel<<<gcd((long long)DD * DD), 256, 0, stream>>>(Wv, Wvt, DD, DD);
    pack_wt_kernel<<<gcd((long long)DD * DD), 256, 0, stream>>>(Wr, Wrt, DD, DD);
    pack_wt_kernel<<<gcd((long long)DD * DD), 256, 0, stream>>>(Wo, Wot, DD, DD);
    pack_wt_kernel<<<gcd((long long)DD * DFF), 256, 0, stream>>>(W1, W1t, DD, DFF);
    pack_wt_kernel<<<gcd((long long)DFF * DD), 256, 0, stream>>>(W2, W2t, DFF, DD);

    // --- cat = [mem_bf16 ; LN1(x)_bf16], positional embedding ---
    mem_to_cat_kernel<<<gcd((long long)BB * MEM * DD), 256, 0, stream>>>(mem, cat);
    ln1_to_cat_kernel<<<dim3(BB * TT), 256, 0, stream>>>(x, ln1g, ln1b, cat);
    pos_emb_kernel<<<gcd((long long)LL * (DD / 2)), 256, 0, stream>>>(pe);

    // --- projections: q, k, v, r ---
    gemm_bf16_wmma<<<dim3(DD / NBLK, TT / MBLK, BB), 256, 0, stream>>>(
        cat + (long long)MEM * DD, DD, (long long)LL * DD,
        Wqt, DD, 0, 1, qf, DD, (long long)TT * DD, DD);
    gemm_bf16_wmma<<<dim3(DD / NBLK, LL / MBLK, BB), 256, 0, stream>>>(
        cat, DD, (long long)LL * DD,
        Wkt, DD, 0, 1, kf, DD, (long long)LL * DD, DD);
    gemm_bf16_wmma<<<dim3(DD / NBLK, LL / MBLK, BB), 256, 0, stream>>>(
        cat, DD, (long long)LL * DD,
        Wvt, DD, 0, 1, vf, DD, (long long)LL * DD, DD);
    gemm_bf16_wmma<<<dim3(DD / NBLK, LL / MBLK, 1), 256, 0, stream>>>(
        pe, DD, 0,
        Wrt, DD, 0, 1, rf, DD, 0, DD);

    // --- repack to head layouts (bf16) ---
    qbias_kernel<<<gcd((long long)BB * TT * DD), 256, 0, stream>>>(qf, bu, bv, qu, qv);
    kv_pack_kernel<<<gcd((long long)BB * LL * DD), 256, 0, stream>>>(kf, vf, kb, vT);
    r_pack_kernel<<<gcd((long long)HH * LL * DH), 256, 0, stream>>>(rf, rb);

    // --- attention scores: content (per b,h) and rel (r shared across batch) ---
    gemm_bf16_wmma<<<dim3(LL / NBLK, TT / MBLK, BB * HH), 256, 0, stream>>>(
        qu, DH, (long long)TT * DH,
        kb, DH, (long long)LL * DH, BB * HH,
        S, LL, (long long)TT * LL, DH);
    gemm_bf16_wmma<<<dim3(LL / NBLK, TT / MBLK, BB * HH), 256, 0, stream>>>(
        qv, DH, (long long)TT * DH,
        rb, DH, (long long)LL * DH, HH,
        R, LL, (long long)TT * LL, DH);

    // --- rel-shift + softmax (writes bf16 attn in-place over S rows) ---
    softmax_shift_kernel<<<dim3(BB * HH * TT), 256, 0, stream>>>(S, R);

    // --- attn @ v -> O [B,H,T,dh] ---
    gemm_bf16_wmma<<<dim3(DH / NBLK, TT / MBLK, BB * HH), 256, 0, stream>>>(
        (const __bf16*)S, 2 * LL, 2LL * TT * LL,
        vT, LL, (long long)DH * LL, BB * HH,
        Oaw, DH, (long long)TT * DH, LL);

    // --- merge heads, output projection, residual + LN2 ---
    merge_heads_kernel<<<gcd((long long)BB * TT * DD), 256, 0, stream>>>(Oaw, AO);
    gemm_bf16_wmma<<<dim3(DD / NBLK, (BB * TT) / MBLK, 1), 256, 0, stream>>>(
        AO, DD, 0, Wot, DD, 0, 1, proj, DD, 0, DD);
    add_ln2_kernel<<<dim3(BB * TT), 256, 0, stream>>>(x, proj, ln2g, ln2b, x2, xf);

    // --- FFN ---
    gemm_bf16_wmma<<<dim3(DFF / NBLK, (BB * TT) / MBLK, 1), 256, 0, stream>>>(
        xf, DD, 0, W1t, DD, 0, 1, ff1, DFF, 0, DD);
    gelu_kernel<<<gcd((long long)BB * TT * DFF), 256, 0, stream>>>(ff1, b1, gb);
    gemm_bf16_wmma<<<dim3(DD / NBLK, (BB * TT) / MBLK, 1), 256, 0, stream>>>(
        gb, DFF, 0, W2t, DFF, 0, 1, ff2, DD, 0, DFF);

    // --- y = x2 + ff + b2 ; new_mem = x ---
    final_kernel<<<gcd((long long)BB * TT * DD), 256, 0, stream>>>(x, x2, ff2, b2, y, nm);
}
